// RelativeNetwork_24635932409876
// MI455X (gfx1250) — matrix operand order repaired and checked by reading, not compiled, not run on hardware
//
#include <hip/hip_runtime.h>
#include <math.h>

typedef __attribute__((ext_vector_type(16))) _Float16 v16h;
typedef __attribute__((ext_vector_type(8)))  _Float16 v8h;
typedef __attribute__((ext_vector_type(8)))  float    v8f;

#define IN_CH 3
#define EMB   8
#define C1    24          // IN_CH * EMB
#define NPOS  256         // 16*16 spatial positions after conv2
#define KPAD  64          // 34 real K (x,y,32ch) padded to 64

// ---------------- 1) embedding lookup:  x(4,3,64,64) int -> e(4,24,64,64) f32
__global__ void rn_embed(const int* __restrict__ x, const float* __restrict__ emb,
                         float* __restrict__ e) {
    int idx = blockIdx.x * 256 + threadIdx.x;          // 49152 = 4*3*4096
    if (idx >= 4 * IN_CH * 4096) return;
    int b   = idx / (IN_CH * 4096);
    int rem = idx - b * (IN_CH * 4096);
    int c   = rem >> 12;
    int pos = rem & 4095;
    int tok = x[idx];
    const float* er = emb + tok * EMB;
#pragma unroll
    for (int k = 0; k < EMB; ++k)
        e[(((b * C1) + c * EMB + k) << 12) + pos] = er[k];
}

// ---------------- 2) conv1 (24->16, 5x5 SAME) + bias + maxpool2 + relu
// grid 64 = (b,oc); 256 thr; each thread 4 pooled pixels of the 32x32 map
__global__ void rn_conv1(const float* __restrict__ e, const float* __restrict__ w1,
                         const float* __restrict__ b1, float* __restrict__ h1) {
    __shared__ float wsh[C1 * 25];
    int b  = blockIdx.x >> 4;
    int oc = blockIdx.x & 15;
    for (int i = threadIdx.x; i < C1 * 25; i += 256) wsh[i] = w1[oc * C1 * 25 + i];
    __syncthreads();
    float bias = b1[oc];
    for (int p = threadIdx.x; p < 1024; p += 256) {
        int ph = p >> 5, pw = p & 31;
        float best = -1e30f;
#pragma unroll
        for (int dy = 0; dy < 2; ++dy)
#pragma unroll
        for (int dx = 0; dx < 2; ++dx) {
            int oh = 2 * ph + dy, ow = 2 * pw + dx;
            float acc = 0.f;
            for (int c = 0; c < C1; ++c) {
                const float* ech = e + (((b * C1) + c) << 12);
                const float* wc  = wsh + c * 25;
#pragma unroll
                for (int kh = 0; kh < 5; ++kh) {
                    int ih = oh + kh - 2;
                    if ((unsigned)ih >= 64u) continue;
                    const float* er = ech + (ih << 6);
                    const float* wr = wc + kh * 5;
#pragma unroll
                    for (int kw = 0; kw < 5; ++kw) {
                        int iw = ow + kw - 2;
                        if ((unsigned)iw < 64u) acc = fmaf(er[iw], wr[kw], acc);
                    }
                }
            }
            best = fmaxf(best, acc);
        }
        h1[(((b << 4) + oc) << 10) + p] = fmaxf(best + bias, 0.f);
    }
}

// ---------------- 3) conv2 (16->32, 5x5 SAME on 32x32) + bias + maxpool2 + relu
// grid 128 = (b,oc); 256 thr; one pooled pixel each (16x16 map)
__global__ void rn_conv2(const float* __restrict__ h1, const float* __restrict__ w2,
                         const float* __restrict__ b2, float* __restrict__ h2) {
    __shared__ float wsh[16 * 25];
    int b  = blockIdx.x >> 5;
    int oc = blockIdx.x & 31;
    for (int i = threadIdx.x; i < 16 * 25; i += 256) wsh[i] = w2[oc * 16 * 25 + i];
    __syncthreads();
    float bias = b2[oc];
    int p  = threadIdx.x;
    int ph = p >> 4, pw = p & 15;
    float best = -1e30f;
#pragma unroll
    for (int dy = 0; dy < 2; ++dy)
#pragma unroll
    for (int dx = 0; dx < 2; ++dx) {
        int oh = 2 * ph + dy, ow = 2 * pw + dx;
        float acc = 0.f;
        for (int c = 0; c < 16; ++c) {
            const float* hch = h1 + (((b << 4) + c) << 10);
            const float* wc  = wsh + c * 25;
#pragma unroll
            for (int kh = 0; kh < 5; ++kh) {
                int ih = oh + kh - 2;
                if ((unsigned)ih >= 32u) continue;
                const float* hr = hch + (ih << 5);
                const float* wr = wc + kh * 5;
#pragma unroll
                for (int kw = 0; kw < 5; ++kw) {
                    int iw = ow + kw - 2;
                    if ((unsigned)iw < 32u) acc = fmaf(hr[iw], wr[kw], acc);
                }
            }
        }
        best = fmaxf(best, acc);
    }
    h2[(((b << 5) + oc) << 8) + p] = fmaxf(best + bias, 0.f);
}

// ---------------- 4) prep f16 operands:
// A[b*256+i][64] = [x_i, y_i, hf[0..31], 0...], Wu[o][64], Wv[o][64] (row = output o)
__global__ void rn_prep(const float* __restrict__ h2, const float* __restrict__ wr,
                        _Float16* __restrict__ A, _Float16* __restrict__ Wu,
                        _Float16* __restrict__ Wv) {
    int tid = blockIdx.x * 256 + threadIdx.x;   // 2048 threads
    if (tid < 1024) {                           // A rows
        int b = tid >> 8, i = tid & 255;
        _Float16* row = A + tid * KPAD;
        row[0] = (_Float16)((float)(i >> 4) * 0.0625f);   // x_i = (i/16)/16
        row[1] = (_Float16)((float)(i & 15) * 0.0625f);   // y_i = (i%16)/16
#pragma unroll
        for (int c = 0; c < 32; ++c)
            row[2 + c] = (_Float16)h2[(((b << 5) + c) << 8) + i];
#pragma unroll
        for (int k = 34; k < KPAD; ++k) row[k] = (_Float16)0.f;
    } else if (tid < 1280) {                    // Wu rows: [wr0, wr1, wr[2..33]]
        int o = tid - 1024;
        _Float16* row = Wu + o * KPAD;
        const float* w = wr + o * 66;
        row[0] = (_Float16)w[0];
        row[1] = (_Float16)w[1];
#pragma unroll
        for (int c = 0; c < 32; ++c) row[2 + c] = (_Float16)w[2 + c];
#pragma unroll
        for (int k = 34; k < KPAD; ++k) row[k] = (_Float16)0.f;
    } else if (tid < 1536) {                    // Wv rows: [-wr0, -wr1, wr[34..65]]
        int o = tid - 1280;
        _Float16* row = Wv + o * KPAD;
        const float* w = wr + o * 66;
        row[0] = (_Float16)(-w[0]);
        row[1] = (_Float16)(-w[1]);
#pragma unroll
        for (int c = 0; c < 32; ++c) row[2 + c] = (_Float16)w[34 + c];
#pragma unroll
        for (int k = 34; k < KPAD; ++k) row[k] = (_Float16)0.f;
    }
}

// ---------------- 5) WMMA GEMM: u = A@Wu^T, v = A@Wv^T, stored transposed
// wave per 16x16 tile: 2048 waves (b, which, i_tile, o_tile), 2 k-steps of 32
__global__ void rn_relgemm(const _Float16* __restrict__ A,
                           const _Float16* __restrict__ Wu,
                           const _Float16* __restrict__ Wv,
                           float* __restrict__ u_t, float* __restrict__ v_t) {
    int wave = (blockIdx.x << 3) + (threadIdx.x >> 5);   // 0..2047
    int lane = threadIdx.x & 31;
    int b     = wave >> 9;
    int rem   = wave & 511;
    int which = rem >> 8;
    int tile  = rem & 255;
    int it = tile >> 4, ot = tile & 15;
    int m  = lane & 15;       // A row / B column (o) per lane
    int hh = lane >> 4;       // lane half selects K sub-block

    const _Float16* arow = A + (((b << 8) + (it << 4) + m) * KPAD);
    const _Float16* wrow = (which ? Wv : Wu) + (((ot << 4) + m) * KPAD);

    v8f acc = {};
#pragma unroll
    for (int ks = 0; ks < 2; ++ks) {
        int kb = ks << 5;
        // A in contiguous k order (matches B): VGPR0-3 -> K = 16*hh + 0..7 ; VGPR4-7 -> K = 16*hh + 8..15
        v8h lo = *(const v8h*)(arow + kb + 16 * hh);
        v8h hi = *(const v8h*)(arow + kb + 16 * hh + 8);
        v16h av;
#pragma unroll
        for (int t = 0; t < 8; ++t) { av[t] = lo[t]; av[t + 8] = hi[t]; }
        // B in contiguous k order (matches A): lanes 0-15 hold K=kb+0..15, lanes 16-31 hold K=kb+16..31
        v16h bv = *(const v16h*)(wrow + kb + 16 * hh);
        acc = __builtin_amdgcn_wmma_f32_16x16x32_f16(
            false, av, false, bv, (short)0, acc, false, false);
    }
    // D 16x16 f32: VGPR r holds row m = r + 8*hh, col n = lane%16 -> transposed store
    float* dst = (which ? v_t : u_t) +
                 ((((b << 8) + (ot << 4) + m) << 8) + (it << 4) + (hh << 3));
#pragma unroll
    for (int r = 0; r < 8; ++r) dst[r] = acc[r];
}

// ---------------- 6) pairwise relu-sum: g[b,o] = mean_ij relu(u[i]+v[j]+br[o])
// grid 1024 = (b*256+o), 256 thr; thread t owns row i=t, loops all j (v in LDS)
__global__ void rn_pairsum(const float* __restrict__ u_t, const float* __restrict__ v_t,
                           const float* __restrict__ br, float* __restrict__ g) {
    __shared__ float vsh[256];
    __shared__ float red[256];
    int row = blockIdx.x;                 // b*256 + o
    int o   = row & 255;
    int tid = threadIdx.x;
    vsh[tid] = v_t[(row << 8) + tid];
    __syncthreads();
    float ui  = u_t[(row << 8) + tid] + br[o];
    float acc = 0.f;
#pragma unroll 8
    for (int j = 0; j < 256; ++j) acc += fmaxf(ui + vsh[j], 0.f);
    red[tid] = acc;
    __syncthreads();
    for (int s = 128; s > 0; s >>= 1) {
        if (tid < s) red[tid] += red[tid + s];
        __syncthreads();
    }
    if (tid == 0) g[row] = red[0] * (1.f / 65536.f);
}

// ---------------- 7) head: FC1+relu, FC2, global mean-sub, per-row L2 norm
__global__ void rn_head(const float* __restrict__ g, const float* __restrict__ wf1,
                        const float* __restrict__ bf1, const float* __restrict__ wf2,
                        const float* __restrict__ bf2, float* __restrict__ out) {
    __shared__ float gs[1024];
    __shared__ float hs[1024];
    __shared__ float o2[256];
    __shared__ float red[256];
    int tid = threadIdx.x;
#pragma unroll
    for (int q = 0; q < 4; ++q) gs[(q << 8) + tid] = g[(q << 8) + tid];
    __syncthreads();
    // FC1: hs[b][tid] = relu(g[b] . wf1[tid] + bf1[tid])
    {
        const float* wrow = wf1 + (tid << 8);
        float bias = bf1[tid];
#pragma unroll
        for (int b = 0; b < 4; ++b) {
            float acc = bias;
            const float* gb = gs + (b << 8);
            for (int k = 0; k < 256; ++k) acc = fmaf(gb[k], wrow[k], acc);
            hs[(b << 8) + tid] = fmaxf(acc, 0.f);
        }
    }
    __syncthreads();
    // FC2: o2[t] for b = t>>6, o = t&63
    {
        int b = tid >> 6, o = tid & 63;
        const float* wrow = wf2 + (o << 8);
        const float* hb   = hs + (b << 8);
        float acc = bf2[o];
        for (int k = 0; k < 256; ++k) acc = fmaf(hb[k], wrow[k], acc);
        o2[tid] = acc;
    }
    __syncthreads();
    // global mean over all 256 outputs
    red[tid] = o2[tid];
    __syncthreads();
    for (int s = 128; s > 0; s >>= 1) {
        if (tid < s) red[tid] += red[tid + s];
        __syncthreads();
    }
    float mean = red[0] * (1.f / 256.f);
    __syncthreads();
    float val = o2[tid] - mean;
    // per-row (64 elems) L2 norm
    red[tid] = val * val;
    __syncthreads();
    for (int s = 32; s > 0; s >>= 1) {
        if ((tid & 63) < s) red[tid] += red[tid + s];
        __syncthreads();
    }
    float nrm = sqrtf(red[(tid >> 6) << 6]);
    out[tid] = val / nrm;
}

extern "C" void kernel_launch(void* const* d_in, const int* in_sizes, int n_in,
                              void* d_out, int out_size, void* d_ws, size_t ws_size,
                              hipStream_t stream) {
    const int*   x   = (const int*)  d_in[0];
    const float* emb = (const float*)d_in[1];
    const float* w1  = (const float*)d_in[2];
    const float* b1  = (const float*)d_in[3];
    const float* w2  = (const float*)d_in[4];
    const float* b2  = (const float*)d_in[5];
    const float* wr  = (const float*)d_in[6];
    const float* br  = (const float*)d_in[7];
    const float* wf1 = (const float*)d_in[8];
    const float* bf1 = (const float*)d_in[9];
    const float* wf2 = (const float*)d_in[10];
    const float* bf2 = (const float*)d_in[11];
    float* out = (float*)d_out;

    char* ws = (char*)d_ws;
    float*    e    = (float*)   (ws + 0);        // 4*24*4096 f32 = 1,572,864 B
    float*    h1   = (float*)   (ws + 1572864);  // 4*16*1024 f32 =   262,144 B
    float*    h2   = (float*)   (ws + 1835008);  // 4*32*256  f32 =   131,072 B
    _Float16* A    = (_Float16*)(ws + 1966080);  // 1024*64 f16   =   131,072 B
    _Float16* Wu   = (_Float16*)(ws + 2097152);  // 256*64 f16    =    32,768 B
    _Float16* Wv   = (_Float16*)(ws + 2129920);  // 256*64 f16    =    32,768 B
    float*    u_t  = (float*)   (ws + 2162688);  // 4*256*256 f32 = 1,048,576 B
    float*    v_t  = (float*)   (ws + 3211264);  // 4*256*256 f32 = 1,048,576 B
    float*    gbuf = (float*)   (ws + 4259840);  // 4*256 f32     =     4,096 B

    rn_embed  <<<192,  256, 0, stream>>>(x, emb, e);
    rn_conv1  <<<64,   256, 0, stream>>>(e, w1, b1, h1);
    rn_conv2  <<<128,  256, 0, stream>>>(h1, w2, b2, h2);
    rn_prep   <<<8,    256, 0, stream>>>(h2, wr, A, Wu, Wv);
    rn_relgemm<<<256,  256, 0, stream>>>(A, Wu, Wv, u_t, v_t);
    rn_pairsum<<<1024, 256, 0, stream>>>(u_t, v_t, br, gbuf);
    rn_head   <<<1,    256, 0, stream>>>(gbuf, wf1, bf1, wf2, bf2, out);
}
